// GNN2_61847529063069
// MI455X (gfx1250) — matrix operand order repaired and checked by reading, not compiled, not run on hardware
//
#include <hip/hip_runtime.h>
#include <hip/hip_bf16.h>

typedef __attribute__((ext_vector_type(16))) _Float16 v16h;
typedef __attribute__((ext_vector_type(8)))  _Float16 v8h;
typedef __attribute__((ext_vector_type(8)))  float    v8f;

#define HEADS  4
#define OUTC   16
#define HID    64
#define NN     128   // nodes per row
#define NEGSL  0.2f

__device__ __forceinline__ float lrelu(float x) { return x > 0.f ? x : NEGSL * x; }

// A-operand fragment: 16x32 f16 tile, rows row0..row0+15, row-major LDS buffer.
// Lane group 0 holds K {0..7,16..23}; group 1 holds K {8..15,24..31}.
__device__ __forceinline__ v16h ldfragA(const _Float16* base, int stride, int row0) {
  const int lane = threadIdx.x & 31;
  const int m = lane & 15, g = lane >> 4;
  const _Float16* p = base + (row0 + m) * stride + 8 * g;
  v8h lo = *(const v8h*)p;
  v8h hi = *(const v8h*)(p + 16);
  v16h r;
#pragma unroll
  for (int t = 0; t < 8; ++t) { r[t] = lo[t]; r[t + 8] = hi[t]; }
  return r;
}

// B-operand fragment: 32x16 f16 tile; lane owns column (row0+lane%16) of B, i.e.
// row (row0+lane%16) of the transposed buffer Bt[col][k]. Lane group 0 holds
// K {0..7,16..23}; group 1 holds K {8..15,24..31}.
__device__ __forceinline__ v16h ldfragB(const _Float16* base, int stride, int row0) {
  const int lane = threadIdx.x & 31;
  const int m = lane & 15, g = lane >> 4;
  const _Float16* p = base + (row0 + m) * stride + 8 * g;
  v8h lo = *(const v8h*)p;
  v8h hi = *(const v8h*)(p + 16);
  v16h r;
#pragma unroll
  for (int t = 0; t < 8; ++t) { r[t] = lo[t]; r[t + 8] = hi[t]; }
  return r;
}

__global__ __launch_bounds__(128) void gat2_kernel(
    const float* __restrict__ xs,  const float* __restrict__ pe,
    const float* __restrict__ w0,  const float* __restrict__ as0,
    const float* __restrict__ ad0, const float* __restrict__ b0,
    const float* __restrict__ w1,  const float* __restrict__ as1,
    const float* __restrict__ ad1, const float* __restrict__ b1,
    const float* __restrict__ fw,  float* __restrict__ out, int R)
{
  __shared__ __align__(16) _Float16 Xbuf[NN * HID];   // node-major activations (A operand)
  __shared__ __align__(16) _Float16 xlT[HID * NN];    // channel-major linear out (B operand)
  __shared__ __align__(16) _Float16 Wlds[HID * HID];  // W rows = Bt rows
  __shared__ __align__(16) _Float16 pT[NN * 32];      // p^T chunk [j][i-slice]
  __shared__ float sArr[HEADS * NN];
  __shared__ float dArr[HEADS * NN];
  __shared__ float smax[HEADS];
  __shared__ float dinv[NN];
  __shared__ float aS[HID], aD[HID], biasL[HID];
  __shared__ float hpart[4][HID];
  __shared__ float hsum[HID];

  const int tid  = threadIdx.x;
  const int wave = tid >> 5;
  const int lane = tid & 31;
  const int lm   = lane & 15, lg = lane >> 4;
  const int pblk = blockIdx.x;
  const int b = pblk / R;
  const int r = pblk % R;

  // ---- build layer-0 input features: [value, pos_enc], K padded 16->32 with zeros
  {
    const int n = tid;  // 128 threads == 128 nodes
    const float* per = pe + (b * NN + n) * 15;
    Xbuf[n * HID + 0] = (_Float16)xs[(b * R + r) * NN + n];
#pragma unroll
    for (int e = 0; e < 15; ++e) Xbuf[n * HID + 1 + e] = (_Float16)per[e];
#pragma unroll
    for (int k = 16; k < 32; ++k) Xbuf[n * HID + k] = (_Float16)0.f;
  }

  for (int layer = 0; layer < 2; ++layer) {
    const float* W  = layer ? (w1 + b * HID * HID) : (w0 + b * HID * 16);
    const float* As = layer ? (as1 + b * HID) : (as0 + b * HID);
    const float* Ad = layer ? (ad1 + b * HID) : (ad0 + b * HID);
    const float* Bi = layer ? (b1 + b * HID) : (b0 + b * HID);
    const int wstride = layer ? 64 : 32;
    const int kchunks = layer ? 2 : 1;

    // ---- stage weights / attention vectors in LDS (f16 weights, K-padded for layer 0)
    if (layer == 0) {
      for (int idx = tid; idx < HID * 32; idx += 128) {
        const int row = idx >> 5, k = idx & 31;
        Wlds[idx] = (k < 16) ? (_Float16)W[row * 16 + k] : (_Float16)0.f;
      }
    } else {
      for (int idx = tid; idx < HID * HID; idx += 128) Wlds[idx] = (_Float16)W[idx];
    }
    if (tid < HID) { aS[tid] = As[tid]; aD[tid] = Ad[tid]; biasL[tid] = Bi[tid]; }
    __syncthreads();

    // ---- linear: xlT[c][n] = sum_f Xbuf[n][f] * W[c][f]   (WMMA, wave owns one n-tile)
    {
      const int ntile = wave;
      for (int mtile = 0; mtile < 8; ++mtile) {
        v8f acc = {};
        for (int kc = 0; kc < kchunks; ++kc) {
          v16h a  = ldfragA(Xbuf + kc * 32, HID, mtile * 16);
          v16h bm = ldfragB(Wlds + kc * 32, wstride, ntile * 16);
          acc = __builtin_amdgcn_wmma_f32_16x16x32_f16(false, a, false, bm,
                                                       (short)0, acc, false, false);
        }
        v8h pk;
#pragma unroll
        for (int v = 0; v < 8; ++v) pk[v] = (_Float16)acc[v];
        // D: lane col = channel, VGPR v = node v+8*lg -> contiguous b128 store into xlT
        *(v8h*)&xlT[(ntile * 16 + lm) * NN + mtile * 16 + 8 * lg] = pk;
      }
    }
    __syncthreads();

    // ---- per-node attention logits s,d
    {
      const int i = tid;
#pragma unroll
      for (int h = 0; h < HEADS; ++h) {
        float s = 0.f, d = 0.f;
#pragma unroll
        for (int c = 0; c < OUTC; ++c) {
          const float v = (float)xlT[(h * OUTC + c) * NN + i];
          s += v * aS[h * OUTC + c];
          d += v * aD[h * OUTC + c];
        }
        sArr[h * NN + i] = s;
        dArr[h * NN + i] = d;
      }
    }
    __syncthreads();
    if (tid < HEADS) {
      float mx = -3.0e38f;
      for (int i = 0; i < NN; ++i) mx = fmaxf(mx, sArr[tid * NN + i]);
      smax[tid] = mx;  // max_i lrelu(s_i+d_j) == lrelu(smax+d_j) (monotone)
    }
    __syncthreads();

    // ---- per-head softmax (unnormalized) + aggregation: out_h = p^T @ XL_h, K chunked by 32
    for (int h = 0; h < HEADS; ++h) {
      const int j = tid;
      const float dj = dArr[h * NN + j];
      const float mj = lrelu(smax[h] + dj);
      float denom = 0.f;
      v8f acc0 = {}, acc1 = {};
      for (int kc = 0; kc < 4; ++kc) {
        __syncthreads();   // prior chunk's WMMA readers of pT are done
#pragma unroll 4
        for (int ii = 0; ii < 32; ++ii) {
          const float e  = lrelu(sArr[h * NN + kc * 32 + ii] + dj);
          const float pv = __expf(e - mj);
          denom += pv;
          pT[j * 32 + ii] = (_Float16)pv;
        }
        __syncthreads();
        v16h bm = ldfragB(xlT + h * OUTC * NN + kc * 32, NN, 0);      // XL_h^T rows
        v16h a0 = ldfragA(pT, 32, (wave * 2) * 16);
        acc0 = __builtin_amdgcn_wmma_f32_16x16x32_f16(false, a0, false, bm,
                                                      (short)0, acc0, false, false);
        v16h a1 = ldfragA(pT, 32, (wave * 2 + 1) * 16);
        acc1 = __builtin_amdgcn_wmma_f32_16x16x32_f16(false, a1, false, bm,
                                                      (short)0, acc1, false, false);
      }
      dinv[j] = 1.f / denom;
      __syncthreads();

      const int kcol = h * OUTC + lm;  // global output channel of this lane
      if (layer == 0) {
#pragma unroll
        for (int t = 0; t < 2; ++t) {
          const v8f& acc = t ? acc1 : acc0;
          const int mtile = wave * 2 + t;
#pragma unroll
          for (int v = 0; v < 8; ++v) {
            const int jj = mtile * 16 + 8 * lg + v;
            const float val = acc[v] * dinv[jj] + biasL[kcol];
            Xbuf[jj * HID + kcol] = (_Float16)val;   // becomes layer-1 input
          }
        }
      } else {
        float partial = 0.f;
#pragma unroll
        for (int t = 0; t < 2; ++t) {
          const v8f& acc = t ? acc1 : acc0;
          const int mtile = wave * 2 + t;
#pragma unroll
          for (int v = 0; v < 8; ++v) {
            const int jj = mtile * 16 + 8 * lg + v;
            partial += acc[v] * dinv[jj] + biasL[kcol];   // node-sum fused here
          }
        }
        partial += __shfl_xor(partial, 16, 32);  // both lane groups hold the same column
        if (lg == 0) hpart[wave][kcol] = partial;
      }
    }
    __syncthreads();
  }

  // ---- final: h[k] = sum_j out2[j][k]; out[o] = sum_k h[k]*final_w[o][k]
  if (tid < HID)
    hsum[tid] = hpart[0][tid] + hpart[1][tid] + hpart[2][tid] + hpart[3][tid];
  __syncthreads();
  if (tid < 2) {
    const float* fwr = fw + b * 2 * HID + tid * HID;
    float o = 0.f;
#pragma unroll
    for (int k = 0; k < HID; ++k) o += hsum[k] * fwr[k];
    out[pblk * 2 + tid] = o;
  }
}

extern "C" void kernel_launch(void* const* d_in, const int* in_sizes, int n_in,
                              void* d_out, int out_size, void* d_ws, size_t ws_size,
                              hipStream_t stream) {
  (void)n_in; (void)d_ws; (void)ws_size; (void)out_size;
  const float* xs  = (const float*)d_in[0];
  const float* pe  = (const float*)d_in[1];
  const float* w0  = (const float*)d_in[2];
  const float* as0 = (const float*)d_in[3];
  const float* ad0 = (const float*)d_in[4];
  const float* b0  = (const float*)d_in[5];
  const float* w1  = (const float*)d_in[6];
  const float* as1 = (const float*)d_in[7];
  const float* ad1 = (const float*)d_in[8];
  const float* b1  = (const float*)d_in[9];
  const float* fw  = (const float*)d_in[10];
  float* out = (float*)d_out;

  const int B = in_sizes[3] / HID;            // att_src0 is [B, H*C]
  const int R = in_sizes[0] / (B * NN);       // xs is [B, R, N]

  gat2_kernel<<<dim3(B * R), dim3(128), 0, stream>>>(
      xs, pe, w0, as0, ad0, b0, w1, as1, ad1, b1, fw, out, R);
}
